// ImpEmbedding_46651934769338
// MI455X (gfx1250) — matrix operand order repaired and checked by reading, not compiled, not run on hardware
//
#include <hip/hip_runtime.h>

// ---- problem constants (from reference) ----
#define BB 1024      // batch (M)
#define VV 50000     // vocab (K)
#define DD 300       // embed dim (N)
#define DP 304       // partial-buffer padded N
#define NL 320       // LDS/weightT padded N (20 tiles of 16)
#define VP 50048     // weightT padded K (multiple of 64, >= 1563*32)
#define EPSN 1e-7f

typedef __attribute__((ext_vector_type(16))) _Float16 v16h;
typedef __attribute__((ext_vector_type(8)))  float    v8f;
typedef __attribute__((ext_vector_type(4)))  _Float16 h4;
typedef __attribute__((ext_vector_type(4)))  unsigned v4u;
typedef __attribute__((ext_vector_type(8)))  int      v8i;
typedef __attribute__((ext_vector_type(4)))  int      v4i;

// Tensor Data Mover path (CDNA5): guarded so a missing builtin falls back to
// the manual b128 LDS copy instead of breaking the build.
#if defined(__HIP_DEVICE_COMPILE__) && \
    __has_builtin(__builtin_amdgcn_tensor_load_to_lds) && \
    __has_builtin(__builtin_amdgcn_s_wait_tensorcnt)
#define USE_TDM 1
#else
#define USE_TDM 0
#endif

// ============================================================================
// One-time: weight (50000x300 fp32) -> weightT (320x50048 f16), transposed,
// zero-padded in both dims. LDS-staged so both global sides are coalesced.
// ============================================================================
__global__ __launch_bounds__(256)
void transpose_weight_kernel(const float* __restrict__ weight,
                             _Float16* __restrict__ wT)
{
    __shared__ __align__(16) _Float16 tile[32][72];
    const int tid = threadIdx.x;
    const int k0  = blockIdx.x * 64;   // 782 tiles -> 50048 exactly
    const int n0  = blockIdx.y * 32;   // 10 tiles  -> 320 exactly

    #pragma unroll
    for (int i = 0; i < 8; ++i) {
        const int e  = tid + i * 256;
        const int nn = e & 31;
        const int kk = e >> 5;
        const int gn = n0 + nn;
        const int gk = k0 + kk;
        float v = 0.f;
        if (gn < DD && gk < VV) v = weight[(size_t)gk * DD + gn];
        tile[nn][kk] = (_Float16)v;
    }
    __syncthreads();

    const int nn  = tid >> 3;
    const int kk0 = (tid & 7) * 8;
    const uint4 val = *(const uint4*)&tile[nn][kk0];
    *(uint4*)(wT + (size_t)(n0 + nn) * VP + k0 + kk0) = val;
}

// ============================================================================
// Stage helpers (force-inlined so LDS address-space inference survives)
// ============================================================================
__device__ __forceinline__ void fill_A_stage(const float* __restrict__ labels,
                                             _Float16 (&A)[64][40],
                                             int m_block, int k, int tid)
{
    #pragma unroll
    for (int i = 0; i < 2; ++i) {
        const int f4  = tid + i * 256;       // 0..511
        const int row = f4 >> 3;             // 0..63
        const int kk  = (f4 & 7) << 2;       // 0,4,..,28
        const bool ok = (k + kk + 4 <= VV);  // only last global step fails
        const float4 v = *(const float4*)(labels +
            (size_t)(m_block + row) * VV + k + (ok ? kk : 0));
        h4 hv;
        hv.x = (_Float16)(ok ? v.x : 0.f);
        hv.y = (_Float16)(ok ? v.y : 0.f);
        hv.z = (_Float16)(ok ? v.z : 0.f);
        hv.w = (_Float16)(ok ? v.w : 0.f);
        *(h4*)&A[row][kk] = hv;              // ds_store_b64
    }
}

#if USE_TDM
// One TDM descriptor moves the whole 32(K) x 320(N) f16 B-tile from weightT
// into LDS, inserting 16B padding after every 64B row -> 80B LDS row stride,
// exactly matching Bsh[n][40]. Issued by wave 0 only; tracked by TENSORcnt.
__device__ __forceinline__ void issue_B_tdm(const _Float16* __restrict__ wT,
                                            _Float16 (&Bb)[NL][40], int k)
{
    const unsigned lds_off = (unsigned)(uintptr_t)&Bb[0][0]; // low32 = LDS byte addr
    const unsigned long long ga = (unsigned long long)(uintptr_t)(wT + k);

    v4u g0;
    g0.x = 1u;                                   // count=1 (valid user D#)
    g0.y = lds_off;                              // lds_addr
    g0.z = (unsigned)ga;                         // global_addr[31:0]
    g0.w = (unsigned)((ga >> 32) & 0x01FFFFFFu)  // global_addr[56:32]
         | (2u << 30);                           // type=2 ("image")

    v8i g1;
    g1[0] = (int)((1u << 16)      // data_size = 2 bytes
                | (1u << 20)      // pad_enable
                | (3u << 22)      // pad_interval: 16 DWORDs (= one 64B row)
                | (3u << 25));    // pad_amount:   4 DWORDs (= 16B)
    g1[1] = (int)(((unsigned)VP & 0xFFFFu) << 16);              // tensor_dim0 lo16
    g1[2] = (int)(((unsigned)VP >> 16) | ((unsigned)NL << 16)); // dim0 hi | dim1 lo
    g1[3] = (int)(32u << 16);     // tensor_dim1 hi (0) | tile_dim0 = 32
    g1[4] = (int)NL;              // tile_dim1 = 320, tile_dim2 = 0
    g1[5] = (int)VP;              // tensor_dim0_stride lo32 (elements)
    g1[6] = 0;                    // stride hi | dim1_stride lo (unused, 2D)
    g1[7] = 0;

    v4i z4 = {0, 0, 0, 0};
#if defined(__clang_major__) && (__clang_major__ >= 23)
    v8i z8 = {0, 0, 0, 0, 0, 0, 0, 0};
    __builtin_amdgcn_tensor_load_to_lds(g0, g1, z4, z4, z8, 0);
#else
    __builtin_amdgcn_tensor_load_to_lds(g0, g1, z4, z4, 0);
#endif
}
#else
__device__ __forceinline__ void fill_B_stage(const _Float16* __restrict__ wT,
                                             _Float16 (&Bb)[NL][40], int k, int tid)
{
    #pragma unroll
    for (int i = 0; i < 5; ++i) {
        const int ch  = tid + i * 256;       // 0..1279
        const int n   = ch >> 2;             // 0..319
        const int off = (ch & 3) << 3;       // 0,8,16,24
        const uint4 g = *(const uint4*)(wT + (size_t)n * VP + k + off);
        *(uint4*)&Bb[n][off] = g;            // ds_store_b128
    }
}
#endif

// ============================================================================
// Fast GEMM: workgroup = 64 rows x 320 cols x one K-chunk; 8 waves =
// 2 (M halves) x 4 (N groups of 80); 10 WMMA / wave / K-step.
// Double-buffered LDS, one barrier per K-step, B via TDM (or b128 copy).
// ============================================================================
__global__ __launch_bounds__(256)
void gemm_fast_kernel(const float* __restrict__ labels,
                      const _Float16* __restrict__ wT,
                      float* __restrict__ part,
                      int steps_per_chunk)
{
    __shared__ __align__(16) _Float16 Ash[2][64][40];
    __shared__ __align__(16) _Float16 Bsh[2][NL][40];

    const int tid    = threadIdx.x;
    const int lane   = tid & 31;
    const int wave   = tid >> 5;
    const int wm     = wave & 1;
    const int wn     = wave >> 1;
    const int mrow   = lane & 15;
    const int halfid = lane >> 4;

    const int m_block = blockIdx.x * 64;
    const int chunk   = blockIdx.y;

    const int steps_total = (VV + 31) / 32;          // 1563
    int s0 = chunk * steps_per_chunk;
    int s1 = s0 + steps_per_chunk;
    if (s1 > steps_total) s1 = steps_total;

    v8f acc[2][5] = {};

    // prologue: stage first K-step into buffer 0
    if (s0 < s1) {
        fill_A_stage(labels, Ash[0], m_block, s0 * 32, tid);
#if USE_TDM
        if (tid < 32) issue_B_tdm(wT, Bsh[0], s0 * 32);
#else
        fill_B_stage(wT, Bsh[0], s0 * 32, tid);
#endif
    }

    for (int s = s0; s < s1; ++s) {
        const int buf = (s - s0) & 1;
#if USE_TDM
        // scalar wait; TENSORcnt is 0 for waves 1..7, so keep it unconditional
        // (avoids exec-mask save/restore in the hot loop)
        __builtin_amdgcn_s_wait_tensorcnt(0);        // B(buf) complete (wave 0)
#endif
        __syncthreads();   // fills of buf visible; prior compute on buf^1 done

        if (s + 1 < s1) {  // prefetch next K-step into the other buffer
            fill_A_stage(labels, Ash[buf ^ 1], m_block, (s + 1) * 32, tid);
#if USE_TDM
            if (tid < 32) issue_B_tdm(wT, Bsh[buf ^ 1], (s + 1) * 32);
#else
            fill_B_stage(wT, Bsh[buf ^ 1], (s + 1) * 32, tid);
#endif
        }

        // ---- fragments + WMMA on buffer `buf` ----
        v16h afrag[2];
        #pragma unroll
        for (int t = 0; t < 2; ++t) {
            const int row = wm * 32 + t * 16 + mrow;
            ((uint4*)&afrag[t])[0] = *(const uint4*)&Ash[buf][row][16 * halfid];
            ((uint4*)&afrag[t])[1] = *(const uint4*)&Ash[buf][row][16 * halfid + 8];
        }
        #pragma unroll
        for (int j = 0; j < 5; ++j) {
            const int col = wn * 80 + j * 16 + mrow;
            v16h bfrag;
            const uint4* q = (const uint4*)&Bsh[buf][col][16 * halfid];
            ((uint4*)&bfrag)[0] = q[0];
            ((uint4*)&bfrag)[1] = q[1];
            #pragma unroll
            for (int t = 0; t < 2; ++t) {
                acc[t][j] = __builtin_amdgcn_wmma_f32_16x16x32_f16(
                    false, afrag[t], false, bfrag, (short)0, acc[t][j],
                    false, false);
            }
        }
    }

    float* pc = part + (size_t)chunk * (BB * DP);
    #pragma unroll
    for (int t = 0; t < 2; ++t) {
        #pragma unroll
        for (int j = 0; j < 5; ++j) {
            const int col = wn * 80 + j * 16 + mrow;
            if (col < DP) {
                const int rowb = m_block + wm * 32 + t * 16 + 8 * halfid;
                float* dst = pc + (size_t)rowb * DP + col;
                #pragma unroll
                for (int r = 0; r < 8; ++r) dst[(size_t)r * DP] = acc[t][j][r];
            }
        }
    }
}

// ---- fallback path (small workspace): inline-convert weight each K-step ----
__global__ __launch_bounds__(256)
void gemm_fallback_kernel(const float* __restrict__ labels,
                          const float* __restrict__ weight,
                          float* __restrict__ part,
                          int steps_per_chunk)
{
    __shared__ __align__(16) _Float16 Ash[64][40];
    __shared__ __align__(16) _Float16 Bsh[NL][40];

    const int tid    = threadIdx.x;
    const int lane   = tid & 31;
    const int wave   = tid >> 5;
    const int wm     = wave & 1;
    const int wn     = wave >> 1;
    const int mrow   = lane & 15;
    const int halfid = lane >> 4;

    const int m_block = blockIdx.x * 64;
    const int chunk   = blockIdx.y;

    const int steps_total = (VV + 31) / 32;
    int s0 = chunk * steps_per_chunk;
    int s1 = s0 + steps_per_chunk;
    if (s1 > steps_total) s1 = steps_total;

    v8f acc[2][5] = {};

    for (int s = s0; s < s1; ++s) {
        const int k = s * 32;

        fill_A_stage(labels, Ash, m_block, k, tid);

        #pragma unroll
        for (int i = 0; i < 10; ++i) {
            const int f4 = tid + i * 256;
            const int kk = f4 / 80;
            const int n4 = f4 % 80;
            const bool okn = (n4 < 75);
            const bool okk = (k + kk < VV);
            const int nc = (okn ? n4 : 74) << 2;
            const int kc = okk ? (k + kk) : (VV - 1);
            const float4 w = *(const float4*)(weight + (size_t)kc * DD + nc);
            const bool ok = okn && okk;
            const int n = n4 << 2;
            Bsh[n + 0][kk] = (_Float16)(ok ? w.x : 0.f);
            Bsh[n + 1][kk] = (_Float16)(ok ? w.y : 0.f);
            Bsh[n + 2][kk] = (_Float16)(ok ? w.z : 0.f);
            Bsh[n + 3][kk] = (_Float16)(ok ? w.w : 0.f);
        }

        __syncthreads();

        v16h afrag[2];
        #pragma unroll
        for (int t = 0; t < 2; ++t) {
            const int row = wm * 32 + t * 16 + mrow;
            ((uint4*)&afrag[t])[0] = *(const uint4*)&Ash[row][16 * halfid];
            ((uint4*)&afrag[t])[1] = *(const uint4*)&Ash[row][16 * halfid + 8];
        }
        #pragma unroll
        for (int j = 0; j < 5; ++j) {
            const int col = wn * 80 + j * 16 + mrow;
            v16h bfrag;
            const uint4* q = (const uint4*)&Bsh[col][16 * halfid];
            ((uint4*)&bfrag)[0] = q[0];
            ((uint4*)&bfrag)[1] = q[1];
            #pragma unroll
            for (int t = 0; t < 2; ++t) {
                acc[t][j] = __builtin_amdgcn_wmma_f32_16x16x32_f16(
                    false, afrag[t], false, bfrag, (short)0, acc[t][j],
                    false, false);
            }
        }

        __syncthreads();
    }

    float* pc = part + (size_t)chunk * (BB * DP);
    #pragma unroll
    for (int t = 0; t < 2; ++t) {
        #pragma unroll
        for (int j = 0; j < 5; ++j) {
            const int col = wn * 80 + j * 16 + mrow;
            if (col < DP) {
                const int rowb = m_block + wm * 32 + t * 16 + 8 * halfid;
                float* dst = pc + (size_t)rowb * DP + col;
                #pragma unroll
                for (int r = 0; r < 8; ++r) dst[(size_t)r * DP] = acc[t][j][r];
            }
        }
    }
}

// ============================================================================
// Reduce K-chunk partials + L2-normalize each row (fixed order: deterministic)
// ============================================================================
__global__ __launch_bounds__(256)
void reduce_norm_kernel(const float* __restrict__ part,
                        float* __restrict__ out,
                        int k_chunks)
{
    const int row = blockIdx.x;
    const int t   = threadIdx.x;
    const int t2  = t + 256;

    float a0 = 0.f, a1 = 0.f;
    for (int c = 0; c < k_chunks; ++c) {
        const float* pc = part + (size_t)c * (BB * DP) + (size_t)row * DP;
        a0 += pc[t];
        if (t2 < DD) a1 += pc[t2];
    }

    __shared__ float red[256];
    red[t] = a0 * a0 + ((t2 < DD) ? a1 * a1 : 0.f);
    __syncthreads();
    #pragma unroll
    for (int s = 128; s > 0; s >>= 1) {
        if (t < s) red[t] += red[t + s];
        __syncthreads();
    }
    const float inv = 1.0f / (sqrtf(red[0]) + EPSN);

    out[(size_t)row * DD + t] = a0 * inv;
    if (t2 < DD) out[(size_t)row * DD + t2] = a1 * inv;
}

// ============================================================================
extern "C" void kernel_launch(void* const* d_in, const int* in_sizes, int n_in,
                              void* d_out, int out_size, void* d_ws, size_t ws_size,
                              hipStream_t stream)
{
    const float* labels = (const float*)d_in[0];   // (1024, 50000) fp32
    const float* weight = (const float*)d_in[1];   // (50000, 300) fp32
    float* out = (float*)d_out;                    // (1024, 300) fp32

    const size_t WT_BYTES  = (size_t)NL * VP * sizeof(_Float16);  // ~30.5 MiB
    const size_t per_chunk = (size_t)BB * DP * sizeof(float);     // ~1.19 MiB
    const int steps_total  = (VV + 31) / 32;                      // 1563

    int kc_fast = 0;
    if (ws_size > WT_BYTES)
        kc_fast = (int)((ws_size - WT_BYTES) / per_chunk);
    if (kc_fast > 10) kc_fast = 10;

    if (kc_fast >= 8) {
        _Float16* wT = (_Float16*)d_ws;
        float* part  = (float*)((char*)d_ws + WT_BYTES);
        int spc = (steps_total + kc_fast - 1) / kc_fast;
        int kc  = (steps_total + spc - 1) / spc;

        transpose_weight_kernel<<<dim3(VP / 64, NL / 32), dim3(256), 0, stream>>>(weight, wT);
        gemm_fast_kernel<<<dim3(BB / 64, kc), dim3(256), 0, stream>>>(labels, wT, part, spc);
        reduce_norm_kernel<<<dim3(BB), dim3(256), 0, stream>>>(part, out, kc);
    } else {
        float* part = (float*)d_ws;
        int kc = (int)(ws_size / per_chunk);
        if (kc < 1)  kc = 1;
        if (kc > 10) kc = 10;
        int spc = (steps_total + kc - 1) / kc;
        kc = (steps_total + spc - 1) / spc;

        gemm_fallback_kernel<<<dim3(BB / 64, kc), dim3(256), 0, stream>>>(labels, weight, part, spc);
        reduce_norm_kernel<<<dim3(BB), dim3(256), 0, stream>>>(part, out, kc);
    }
}
